// Gpt2Attention_22900765622610
// MI455X (gfx1250) — matrix operand order repaired and checked by reading, not compiled, not run on hardware
//
#include <hip/hip_runtime.h>

// Problem constants
#define Bb 4
#define Ss 2048
#define Ee 1024
#define Hh 16
#define Dd 64
#define N_QKV 3072

typedef __attribute__((ext_vector_type(16))) __bf16 bf16x16;
typedef __attribute__((ext_vector_type(8)))  float  floatx8;

union Frag { unsigned int u[8]; bf16x16 v; };

__device__ __forceinline__ unsigned short f32_to_bf16(float f) {
  union { float f; unsigned int u; } x; x.f = f;
  unsigned int r = x.u + 0x7FFFu + ((x.u >> 16) & 1u);  // round-to-nearest-even
  return (unsigned short)(r >> 16);
}

// A-operand (16x32 bf16) per-lane K index for VGPR v, lane-half (contiguous k = 16*half + 2*v, matches B fills)
__device__ __forceinline__ int kmapA(int v, int half) {
  return 2 * v + 16 * half;
}

__device__ __forceinline__ floatx8 wmma_bf16(const Frag& a, const Frag& b, floatx8 c) {
  return __builtin_amdgcn_wmma_f32_16x16x32_bf16(false, a.v, false, b.v,
                                                 (short)0, c, false, false);
}

// CDNA5 async global->LDS copy (ASYNCcnt path, CDNA5 ISA)
__device__ __forceinline__ void async_load_b128(unsigned lds_off, const void* g) {
  unsigned long long ga = (unsigned long long)g;
  asm volatile("global_load_async_to_lds_b128 %0, %1, off"
               :: "v"(lds_off), "v"(ga) : "memory");
}
__device__ __forceinline__ void wait_async0() {
  asm volatile("s_wait_asynccnt 0x0" ::: "memory");
}

// 16-lane max reduction as a DPP16 butterfly (no LDS round-trips):
// quad_perm[1,0,3,2], quad_perm[2,3,0,1], row_half_mirror, row_mirror
template<int CTRL>
__device__ __forceinline__ float dpp_maxstep(float x) {
  float y = __int_as_float(__builtin_amdgcn_update_dpp(
      0, __float_as_int(x), CTRL, 0xF, 0xF, true));
  return fmaxf(x, y);
}
__device__ __forceinline__ float rowmax16(float x) {
  x = dpp_maxstep<0xB1>(x);    // quad_perm 1,0,3,2
  x = dpp_maxstep<0x4E>(x);    // quad_perm 2,3,0,1
  x = dpp_maxstep<0x141>(x);   // row_half_mirror
  x = dpp_maxstep<0x140>(x);   // row_mirror
  return x;
}

// ---------------------------------------------------------------------------
// Elementwise fp32 -> bf16 (vectorized), one HBM pass
// ---------------------------------------------------------------------------
__global__ __launch_bounds__(256)
void cvt_bf16(const float* __restrict__ in, unsigned short* __restrict__ out, int n4) {
  int i = (blockIdx.x * 256 + threadIdx.x);
  if (i >= n4) return;
  float4 f = *(const float4*)(in + (size_t)i * 4);
  uint2 p;
  p.x = (unsigned)f32_to_bf16(f.x) | ((unsigned)f32_to_bf16(f.y) << 16);
  p.y = (unsigned)f32_to_bf16(f.z) | ((unsigned)f32_to_bf16(f.w) << 16);
  *(uint2*)(out + (size_t)i * 4) = p;
}

// ---------------------------------------------------------------------------
// fp32 W[K][N] -> bf16 Wt[N][K] (tiled transpose through LDS), one-time pass
// ---------------------------------------------------------------------------
__global__ __launch_bounds__(256)
void cvt_transpose(const float* __restrict__ W, unsigned short* __restrict__ Wt,
                   int K, int N) {
  __shared__ unsigned short t[32][33];
  int n0 = blockIdx.x * 32, k0 = blockIdx.y * 32;
  int tx = threadIdx.x & 31, ty = threadIdx.x >> 5;  // 32 x 8
#pragma unroll
  for (int j = 0; j < 32; j += 8)
    t[ty + j][tx] = f32_to_bf16(W[(size_t)(k0 + ty + j) * N + n0 + tx]);
  __syncthreads();
#pragma unroll
  for (int j = 0; j < 32; j += 8)
    Wt[(size_t)(n0 + ty + j) * K + k0 + tx] = t[tx][ty + j];
}

// ---------------------------------------------------------------------------
// bf16 WMMA GEMM, 128x128x32 tiles, 8 waves, double-buffered async-LDS staging.
// A: bf16 [M][K] row-major.  Bt: bf16 [N][K] (pre-transposed weights).
// MODE 0: epilogue +b_qkv, scatter q[B,H,S,D], k[B,H,S,D], vT[B,H,D,S] bf16.
// MODE 1: epilogue +b_proj, fp32 store.
// ---------------------------------------------------------------------------
#define LDSTRIDE 40  // shorts; 80B rows = 5x16B (keeps B128 alignment, conflict-free)

template<int MODE>
__global__ __launch_bounds__(256)
void wmma_gemm(const unsigned short* __restrict__ A,
               const unsigned short* __restrict__ Bt,
               int M, int N, int K,
               const float* __restrict__ bias,
               unsigned short* __restrict__ qo,
               unsigned short* __restrict__ ko,
               unsigned short* __restrict__ vto,
               float* __restrict__ out) {
  __shared__ __align__(16) unsigned short As[2][128][LDSTRIDE];
  __shared__ __align__(16) unsigned short Bs[2][128][LDSTRIDE];
  (void)M; (void)N;

  const int tid  = threadIdx.x;
  const int lane = tid & 31;
  const int wave = tid >> 5;
  const int wm   = wave & 3;          // 4 waves along M (32 rows each)
  const int wn   = wave >> 2;         // 2 waves along N (64 cols each)
  const int half = lane >> 4;
  const int nl   = lane & 15;

  const int gm0 = blockIdx.y * 128;
  const int gn0 = blockIdx.x * 128;

  const unsigned asBase = (unsigned)(size_t)&As[0][0][0];
  const unsigned bsBase = (unsigned)(size_t)&Bs[0][0][0];
  const unsigned bufBytes = 128 * LDSTRIDE * 2;

  // stage one 128x32 bf16 tile (512 x 16B chunks) via async b128 copies
  auto stage = [&](const unsigned short* gbase, unsigned ldsBase) {
#pragma unroll
    for (int i = 0; i < 2; ++i) {
      int c = tid + i * 256;               // 0..511
      int row = c >> 2, kc = c & 3;
      async_load_b128(ldsBase + (unsigned)(row * (LDSTRIDE * 2) + kc * 16),
                      gbase + (size_t)row * K + kc * 8);
    }
  };

  floatx8 acc[2][4];
  floatx8 z = {};
#pragma unroll
  for (int t = 0; t < 2; ++t)
#pragma unroll
    for (int j = 0; j < 4; ++j) acc[t][j] = z;

  const int nk = K >> 5;
  stage(A + (size_t)gm0 * K, asBase);
  stage(Bt + (size_t)gn0 * K, bsBase);

  for (int kk = 0; kk < nk; ++kk) {
    wait_async0();
    __syncthreads();                       // tile kk resident; prior reads done
    int buf = kk & 1;
    if (kk + 1 < nk) {
      int nb = (kk + 1) & 1;
      stage(A + (size_t)gm0 * K + (kk + 1) * 32, asBase + nb * bufBytes);
      stage(Bt + (size_t)gn0 * K + (kk + 1) * 32, bsBase + nb * bufBytes);
    }

    Frag af[2];
#pragma unroll
    for (int t = 0; t < 2; ++t) {
      int m = wm * 32 + t * 16 + nl;
#pragma unroll
      for (int v = 0; v < 8; ++v)
        af[t].u[v] = *(const unsigned int*)&As[buf][m][kmapA(v, half)];
    }
    Frag bfr[4];
#pragma unroll
    for (int j = 0; j < 4; ++j) {
      int n = wn * 64 + j * 16 + nl;
#pragma unroll
      for (int v = 0; v < 8; ++v)
        bfr[j].u[v] = *(const unsigned int*)&Bs[buf][n][2 * v + 16 * half];
    }
#pragma unroll
    for (int t = 0; t < 2; ++t)
#pragma unroll
      for (int j = 0; j < 4; ++j)
        acc[t][j] = wmma_bf16(af[t], bfr[j], acc[t][j]);
  }

  // ---- epilogue ----
#pragma unroll
  for (int t = 0; t < 2; ++t) {
#pragma unroll
    for (int j = 0; j < 4; ++j) {
#pragma unroll
      for (int r = 0; r < 8; ++r) {
        int gm = gm0 + wm * 32 + t * 16 + r + 8 * half;
        int gn = gn0 + wn * 64 + j * 16 + nl;
        float val = acc[t][j][r] + bias[gn];
        if (MODE == 0) {
          int tt = gn >> 10, h = (gn >> 6) & 15, d = gn & 63;
          int b = gm >> 11, s = gm & 2047;
          unsigned short bits = f32_to_bf16(val);
          size_t bh = (size_t)b * Hh + h;
          if (tt == 0)      qo[(bh * Ss + s) * Dd + d] = bits;
          else if (tt == 1) ko[(bh * Ss + s) * Dd + d] = bits;
          else              vto[(bh * Dd + d) * Ss + s] = bits;  // V transposed
        } else {
          out[(size_t)gm * Ee + gn] = val;
        }
      }
    }
  }
}

// ---------------------------------------------------------------------------
// Flash attention tile step (MASKED only on the causal boundary tiles).
// Row max: DPP16 butterfly.  Row sum: WMMA with all-ones B operand (gives the
// denominator replicated across lanes in C-layout, same bf16 P as numerator).
// ---------------------------------------------------------------------------
template<bool MASKED>
__device__ __forceinline__ void attn_step(
    int key0, int m0, int half, int nl, float scale,
    const unsigned short* __restrict__ Kb, const unsigned short* __restrict__ Vb,
    const Frag (&qf)[2], floatx8 (&o)[4], float (&mrow)[8], float (&lrow)[8],
    unsigned short (*Ps)[36]) {
  floatx8 z = {};
  Frag kf[2][2];
#pragma unroll
  for (int j = 0; j < 2; ++j) {
    const unsigned short* kr = Kb + (size_t)(key0 + j * 16 + nl) * Dd;
#pragma unroll
    for (int i = 0; i < 2; ++i)
#pragma unroll
      for (int v = 0; v < 8; ++v)
        kf[i][j].u[v] = *(const unsigned int*)(kr + i * 32 + 2 * v + 16 * half);
  }
  floatx8 s0 = z, s1 = z;
  s0 = wmma_bf16(qf[0], kf[0][0], s0);
  s0 = wmma_bf16(qf[1], kf[1][0], s0);
  s1 = wmma_bf16(qf[0], kf[0][1], s1);
  s1 = wmma_bf16(qf[1], kf[1][1], s1);

  float alpha[8], p0[8], p1[8];
#pragma unroll
  for (int r = 0; r < 8; ++r) {
    float va = s0[r] * scale;
    float vb = s1[r] * scale;
    if (MASKED) {
      int query = m0 + r + 8 * half;
      if (key0 + nl      > query) va = -1e30f;
      if (key0 + 16 + nl > query) vb = -1e30f;
    }
    float rm = rowmax16(fmaxf(va, vb));
    float nm = fmaxf(mrow[r], rm);
    alpha[r] = __expf(mrow[r] - nm);
    p0[r] = __expf(va - nm);
    p1[r] = __expf(vb - nm);
    mrow[r] = nm;
  }
#pragma unroll
  for (int j = 0; j < 4; ++j)
#pragma unroll
    for (int r = 0; r < 8; ++r) o[j][r] *= alpha[r];

  // P: C-layout -> LDS -> A-layout fragment (per-wave region, in-order DS)
#pragma unroll
  for (int r = 0; r < 8; ++r) {
    Ps[r + 8 * half][nl]      = f32_to_bf16(p0[r]);
    Ps[r + 8 * half][16 + nl] = f32_to_bf16(p1[r]);
  }
  Frag pf;
#pragma unroll
  for (int v = 0; v < 8; ++v)
    pf.u[v] = *(const unsigned int*)&Ps[nl][kmapA(v, half)];

  // denominator: rowsum(P) = P x ones, replicated over lanes in C-layout
  Frag ones;
#pragma unroll
  for (int v = 0; v < 8; ++v) ones.u[v] = 0x3F803F80u;  // bf16 1.0 pairs
  floatx8 rs = wmma_bf16(pf, ones, z);
#pragma unroll
  for (int r = 0; r < 8; ++r) lrow[r] = lrow[r] * alpha[r] + rs[r];

#pragma unroll
  for (int j = 0; j < 4; ++j) {
    Frag vf;
    const unsigned short* vr = Vb + (size_t)(j * 16 + nl) * Ss + key0;
#pragma unroll
    for (int v = 0; v < 8; ++v)
      vf.u[v] = *(const unsigned int*)(vr + 2 * v + 16 * half);
    o[j] = wmma_bf16(pf, vf, o[j]);
  }
}

// grid (S/64, H, B), 128 threads = 4 waves, one 16-query tile per wave
__global__ __launch_bounds__(128)
void flash_attn(const unsigned short* __restrict__ Q,
                const unsigned short* __restrict__ Kg,
                const unsigned short* __restrict__ VT,
                unsigned short* __restrict__ O,
                const int* __restrict__ layer_idx) {
  __shared__ unsigned short Ps[4][16][36];

  const int lane = threadIdx.x & 31;
  const int wave = threadIdx.x >> 5;
  const int half = lane >> 4;
  const int nl   = lane & 15;
  const int h = blockIdx.y, b = blockIdx.z;
  const int m0 = (blockIdx.x * 4 + wave) * 16;

  const size_t bh = (size_t)b * Hh + h;
  const unsigned short* Qb = Q  + bh * Ss * Dd;
  const unsigned short* Kb = Kg + bh * Ss * Dd;
  const unsigned short* Vb = VT + bh * Dd * Ss;

  const float scale = 0.125f / (float)(*layer_idx + 1);  // rsqrt(64)/(l+1)

  Frag qf[2];
#pragma unroll
  for (int f = 0; f < 2; ++f) {
    const unsigned short* qr = Qb + (size_t)(m0 + nl) * Dd + 32 * f;
#pragma unroll
    for (int v = 0; v < 8; ++v)
      qf[f].u[v] = *(const unsigned int*)(qr + kmapA(v, half));
  }

  floatx8 o[4];
  floatx8 z = {};
#pragma unroll
  for (int j = 0; j < 4; ++j) o[j] = z;
  float mrow[8], lrow[8];
#pragma unroll
  for (int r = 0; r < 8; ++r) { mrow[r] = -3.0e38f; lrow[r] = 0.f; }

  const int nfull = (m0 + 1) >> 5;   // tiles with key0+31 <= m0: no masking
  const int nkt   = (m0 + 47) >> 5;  // total tiles (keys up to m0+15)
  for (int kt = 0; kt < nfull; ++kt)
    attn_step<false>(kt * 32, m0, half, nl, scale, Kb, Vb, qf, o, mrow, lrow, Ps[wave]);
  for (int kt = nfull; kt < nkt; ++kt)
    attn_step<true>(kt * 32, m0, half, nl, scale, Kb, Vb, qf, o, mrow, lrow, Ps[wave]);

  float inv[8];
#pragma unroll
  for (int r = 0; r < 8; ++r) inv[r] = 1.0f / lrow[r];
#pragma unroll
  for (int j = 0; j < 4; ++j)
#pragma unroll
    for (int r = 0; r < 8; ++r) {
      int s = m0 + r + 8 * half;
      int d = j * 16 + nl;
      O[((size_t)b * Ss + s) * (Hh * Dd) + h * Dd + d] = f32_to_bf16(o[j][r] * inv[r]);
    }
}

extern "C" void kernel_launch(void* const* d_in, const int* in_sizes, int n_in,
                              void* d_out, int out_size, void* d_ws, size_t ws_size,
                              hipStream_t stream) {
  (void)in_sizes; (void)n_in; (void)out_size; (void)ws_size;
  const float* hidden  = (const float*)d_in[0];
  // d_in[1] = mask: causal tril, applied analytically in-kernel
  const float* W_qkv   = (const float*)d_in[2];
  const float* b_qkv   = (const float*)d_in[3];
  const float* W_proj  = (const float*)d_in[4];
  const float* b_proj  = (const float*)d_in[5];
  const int*   layeri  = (const int*)d_in[6];
  float* out = (float*)d_out;

  const size_t NH = (size_t)Bb * Ss * Ee;       // 8,388,608 (hidden / per-tensor)
  unsigned short* hbf  = (unsigned short*)d_ws; // hidden bf16 [8192][1024]
  unsigned short* wqt  = hbf + NH;              // W_qkv^T bf16 [3072][1024]
  unsigned short* wpt  = wqt + (size_t)N_QKV * Ee; // W_proj^T bf16 [1024][1024]
  unsigned short* qb   = wpt + (size_t)Ee * Ee;
  unsigned short* kb   = qb + NH;
  unsigned short* vtb  = kb + NH;
  unsigned short* ob   = vtb + NH;              // attn out bf16 [8192][1024]

  // 0) one-pass fp32->bf16 conversions (+ weight transposes to [N][K])
  cvt_bf16<<<dim3((unsigned)(NH / 4 / 256)), dim3(256), 0, stream>>>(hidden, hbf, (int)(NH / 4));
  cvt_transpose<<<dim3(N_QKV / 32, Ee / 32), dim3(256), 0, stream>>>(W_qkv, wqt, Ee, N_QKV);
  cvt_transpose<<<dim3(Ee / 32, Ee / 32), dim3(256), 0, stream>>>(W_proj, wpt, Ee, Ee);

  // 1) QKV projection GEMM (async-LDS staged bf16 WMMA), scatter q/k/vT
  wmma_gemm<0><<<dim3(N_QKV / 128, (Bb * Ss) / 128), dim3(256), 0, stream>>>(
      hbf, wqt, Bb * Ss, N_QKV, Ee, b_qkv, qb, kb, vtb, nullptr);

  // 2) causal flash attention
  flash_attn<<<dim3(Ss / 64, Hh, Bb), dim3(128), 0, stream>>>(qb, kb, vtb, ob, layeri);

  // 3) output projection GEMM, fp32 out + bias
  wmma_gemm<1><<<dim3(Ee / 128, (Bb * Ss) / 128), dim3(256), 0, stream>>>(
      ob, wpt, Bb * Ss, Ee, Ee, b_proj, nullptr, nullptr, nullptr, out);
}
